// ConvBlock_61100204753642
// MI455X (gfx1250) — matrix operand order repaired and checked by reading, not compiled, not run on hardware
//
#include <hip/hip_runtime.h>
#include <hip/hip_bf16.h>
#include <math.h>

typedef __attribute__((ext_vector_type(16))) _Float16 v16h;
typedef __attribute__((ext_vector_type(8)))  _Float16 v8h;
typedef __attribute__((ext_vector_type(4)))  _Float16 v4h;
typedef __attribute__((ext_vector_type(8)))  float    v8f;

#define IH   128
#define IW   128
#define CIN  128
#define COUT 256
#define NBATCH 16
#define KTOT 1152          // 9 taps * 128 in-channels
#define NKC  36            // K chunks of 32 (chunk-aligned: 128 = 4*32 per tap)
#define ASTRIDE 40         // padded LDS row stride in halves (80B: conflict-free b128)
#define CHW (32L * IH * IW)   // ci-chunk stride in xq elements

// ---------------------------------------------------------------- absmax(x)
__global__ void k_absmax(const float4* __restrict__ x4, unsigned* __restrict__ maxbits,
                         long n4) {
    __shared__ unsigned sm[256];
    unsigned m = 0u;
    for (long i = (long)blockIdx.x * blockDim.x + threadIdx.x; i < n4;
         i += (long)gridDim.x * blockDim.x) {
        float4 v = x4[i];
        m = max(m, __float_as_uint(fabsf(v.x)));
        m = max(m, __float_as_uint(fabsf(v.y)));
        m = max(m, __float_as_uint(fabsf(v.z)));
        m = max(m, __float_as_uint(fabsf(v.w)));
    }
    sm[threadIdx.x] = m;
    __syncthreads();
    for (int s = 128; s > 0; s >>= 1) {
        if ((int)threadIdx.x < s) sm[threadIdx.x] = max(sm[threadIdx.x], sm[threadIdx.x + s]);
        __syncthreads();
    }
    if (threadIdx.x == 0) atomicMax(maxbits, sm[0]);
}

// ------------------------------------------- quantize W + pack B fragments
// Fragment (g = o/16, kc) = register image of 32K x 16N B-tile:
// lane l<16: N=l, K-local 0..15 ; lane l>=16: N=l-16, K-local 16..31.
__global__ void k_quantW(const float* __restrict__ W, float* __restrict__ fw,
                         _Float16* __restrict__ wq) {
    int o = threadIdx.x;                       // one thread per output channel
    const float* wo = W + (long)o * KTOT;      // OIHW: idx = o*1152 + ci*9 + p
    float s = 0.f;
    for (int k = 0; k < KTOT; ++k) s += fabsf(wo[k]);
    if (s == 0.f) s = 1.f;
    const double factor = 4096.0;                       // sqrt(2^24)
    const double sf     = sqrt(48.0 / 1152.0);
    float f = (float)((factor / sf - sqrt(1152.0 / 12.0) * 5.0) / (double)s);
    fw[o] = f;
    int g = o >> 4, r = o & 15;
    for (int kc = 0; kc < NKC; ++kc) {
        int p  = kc >> 2;                 // tap index (kh*3+kw)
        int cb = (kc & 3) << 5;           // ci base of this chunk
        _Float16* frag = wq + ((long)(g * NKC + kc) << 9);   // 512 halves/frag
        for (int j2 = 0; j2 < 32; ++j2) {
            int ci = cb + j2;
            float v = rintf(wo[ci * 9 + p] * f);             // round-half-even
            int lane = r + ((j2 >> 4) << 4);
            frag[lane * 16 + (j2 & 15)] = (_Float16)v;
        }
    }
}

// ---------------------------------------------------------------- quantize x
__global__ void k_quantX(const float4* __restrict__ x4, v4h* __restrict__ xq4,
                         const unsigned* __restrict__ maxbits, long n4) {
    float xm = __uint_as_float(*maxbits);
    float fx = xm > 0.f ? (float)(4096.0 * sqrt(48.0 / 1152.0) - 0.5) / xm : 1.f;
    for (long i = (long)blockIdx.x * blockDim.x + threadIdx.x; i < n4;
         i += (long)gridDim.x * blockDim.x) {
        float4 v = x4[i];
        v4h h;
        h[0] = (_Float16)rintf(fx * v.x);
        h[1] = (_Float16)rintf(fx * v.y);
        h[2] = (_Float16)rintf(fx * v.z);
        h[3] = (_Float16)rintf(fx * v.w);
        xq4[i] = h;
    }
}

// -------------------------------------------------- implicit-GEMM conv core
// Block tile: 64 pixels (one n, oh, half row) x 256 channels. 8 waves:
// wn = wid&3 (64-channel column), wm = wid>>2 (32-pixel row). Each wave:
// 2 A-frags x 4 B-frags -> 8 accumulators, 8 WMMAs per 32-K step.
__global__ __launch_bounds__(256)
void k_conv(const _Float16* __restrict__ xq, const _Float16* __restrict__ wq,
            const float* __restrict__ fw, const float* __restrict__ bias,
            const unsigned* __restrict__ maxbits, float* __restrict__ out) {
    union SMem {
        _Float16 a[2][64 * ASTRIDE];       // double-buffered A tile [m][ci]
        float    c[COUT * 65];             // epilogue C transpose [N][m]
    };
    __shared__ SMem sm;

    int wg  = blockIdx.x;                  // 4096 blocks
    int n   = wg >> 8;                     // 256 tiles per image (128 oh * 2)
    int rem = wg & 255;
    int oh  = rem >> 1;
    int ow0 = (rem & 1) << 6;              // 0 or 64

    int t    = threadIdx.x;
    int lane = t & 31;
    int wid  = t >> 5;
    int wn   = wid & 3;
    int wm   = wid >> 2;

    v8f acc[2][4] = {};

    // A-stage loader: thread -> (ci = t>>3, 8 consecutive m = (t&7)*8)
    int lci = t >> 3;
    int lm  = (t & 7) << 3;
    const _Float16* tb = xq + (((long)n * CIN + lci) * IH + oh) * IW + ow0 + lm;

    // A-fragment LDS offsets (contiguous k, same as wq: lane<16 K{0..15}, lane>=16 K{16..31})
    int acb = (lane >> 4) << 4;
    int ar0 = (wm * 32 + (lane & 15)) * ASTRIDE + acb;
    int ar1 = ar0 + 16 * ASTRIDE;

    // B-fragment base pointers (lane offset folded in)
    const _Float16* bp[4];
    #pragma unroll
    for (int f = 0; f < 4; ++f)
        bp[f] = wq + (((long)(wn * 4 + f) * NKC) << 9) + lane * 16;

    int buf = 0;
    int kh = 0, kw = 0;
    #pragma unroll 1
    for (int p = 0; p < 9; ++p) {
        int ih = oh + kh - 1;
        if ((unsigned)ih < IH) {           // uniform: whole tap is zero otherwise
            int soff = (kh - 1) * IW + (kw - 1);   // scalar gather offset
            int biw  = ow0 + lm + kw - 1;          // thread iw base

            // gather A for all 4 ci-chunks up front (branchless, select-padded;
            // +/-1-element overreads stay inside the workspace slab)
            _Float16 v[4][8];
            #pragma unroll
            for (int cc = 0; cc < 4; ++cc) {
                const _Float16* src = tb + soff + cc * CHW;
                #pragma unroll
                for (int i = 0; i < 8; ++i) {
                    _Float16 xv = src[i];
                    v[cc][i] = ((unsigned)(biw + i) < IW) ? xv : (_Float16)0.f;
                }
            }

            #pragma unroll
            for (int cc = 0; cc < 4; ++cc) {
                // B fragments: coalesced 32B/lane loads, offset folds to imm
                v16h bf[4];
                #pragma unroll
                for (int f = 0; f < 4; ++f)
                    bf[f] = *(const v16h*)(bp[f] + (((p << 2) + cc) << 9));

                _Float16* ab = sm.a[buf];
                #pragma unroll
                for (int i = 0; i < 8; ++i)
                    ab[(lm + i) * ASTRIDE + lci] = v[cc][i];   // transposed store
                __syncthreads();

                v8h lo0 = *(const v8h*)(ab + ar0);
                v8h hi0 = *(const v8h*)(ab + ar0 + 8);
                v8h lo1 = *(const v8h*)(ab + ar1);
                v8h hi1 = *(const v8h*)(ab + ar1 + 8);
                v16h a0 = __builtin_shufflevector(lo0, hi0, 0, 1, 2, 3, 4, 5, 6, 7,
                                                  8, 9, 10, 11, 12, 13, 14, 15);
                v16h a1 = __builtin_shufflevector(lo1, hi1, 0, 1, 2, 3, 4, 5, 6, 7,
                                                  8, 9, 10, 11, 12, 13, 14, 15);
                #pragma unroll
                for (int f = 0; f < 4; ++f) {
                    acc[0][f] = __builtin_amdgcn_wmma_f32_16x16x32_f16(
                        false, a0, false, bf[f], (short)0, acc[0][f], false, false);
                    acc[1][f] = __builtin_amdgcn_wmma_f32_16x16x32_f16(
                        false, a1, false, bf[f], (short)0, acc[1][f], false, false);
                }
                buf ^= 1;                   // double buffer: 1 barrier / K-step
            }
        }
        if (++kw == 3) { kw = 0; ++kh; }
    }

    // C -> LDS transpose (f32 C layout: N=lane%16, M=v+8*(lane>=16))
    __syncthreads();
    #pragma unroll
    for (int a = 0; a < 2; ++a)
        #pragma unroll
        for (int f = 0; f < 4; ++f) {
            int N  = wn * 64 + f * 16 + (lane & 15);
            int Mb = wm * 32 + a * 16 + ((lane >> 4) << 3);
            #pragma unroll
            for (int vv = 0; vv < 8; ++vv)
                sm.c[N * 65 + Mb + vv] = acc[a][f][vv];
        }
    __syncthreads();

    // dequant + bias + ReLU; lane-contiguous 128B stores
    float xm = __uint_as_float(*maxbits);
    float fx = xm > 0.f ? (float)(4096.0 * sqrt(48.0 / 1152.0) - 0.5) / xm : 1.f;
    long outBase = (((long)n * COUT) * IH + oh) * IW + ow0;
    for (int i = 0; i < 32; ++i) {
        int o = wid * 32 + i;                      // wave-uniform channel
        float s  = 1.f / (fx * fw[o]);
        float bv = bias[o];
        float r0 = sm.c[o * 65 + lane] * s + bv;
        float r1 = sm.c[o * 65 + 32 + lane] * s + bv;
        r0 = r0 > 0.f ? r0 : 0.f;
        r1 = r1 > 0.f ? r1 : 0.f;
        long ob = outBase + (long)o * (IH * IW);
        out[ob + lane]      = r0;
        out[ob + 32 + lane] = r1;
    }
}

extern "C" void kernel_launch(void* const* d_in, const int* in_sizes, int n_in,
                              void* d_out, int out_size, void* d_ws, size_t ws_size,
                              hipStream_t stream) {
    const float* x = (const float*)d_in[0];   // [16,128,128,128]
    const float* W = (const float*)d_in[1];   // [256,128,3,3]
    const float* b = (const float*)d_in[2];   // [256]
    float* out = (float*)d_out;

    char* ws = (char*)d_ws;
    unsigned*  maxbits = (unsigned*)ws;                 // 64B scalar slab
    float*     fw      = (float*)(ws + 64);             // 256 f32
    _Float16*  wq      = (_Float16*)(ws + 2048);        // 16*36*512 halves (576KB)
    _Float16*  xq      = (_Float16*)(ws + (1u << 20));  // 33.5M halves (64MB +2B slack)

    const long nx4 = (long)NBATCH * CIN * IH * IW / 4;  // 8,388,608 float4s

    hipMemsetAsync(ws, 0, 64, stream);
    k_absmax<<<2048, 256, 0, stream>>>((const float4*)x, maxbits, nx4);
    k_quantW<<<1, 256, 0, stream>>>(W, fw, wq);
    k_quantX<<<8192, 256, 0, stream>>>((const float4*)x, (v4h*)xq, maxbits, nx4);
    k_conv<<<NBATCH * IH * (IW / 64), 256, 0, stream>>>(xq, wq, fw, b, maxbits, out);
}
